// MambaModel_50259707298216
// MI455X (gfx1250) — matrix operand order repaired and checked by reading, not compiled, not run on hardware
//
#include <hip/hip_runtime.h>
#include <hip/hip_bf16.h>

// ---------------- model dims ----------------
#define NTOKEN   60697
#define D_MODEL  512
#define NLAYERS  4
#define D_STATE  16
#define D_CONV   4
#define D_INNER  1024           // EXPAND * D_MODEL
#define DT_RANK  32             // (512+15)/16
#define B_SZ     4
#define SEQ      2048
#define ROWS     (B_SZ * SEQ)   // 8192
#define XZ_N     (2 * D_INNER)  // 2048
#define XDBL_N   (DT_RANK + 2 * D_STATE) // 64
#define LN_EPS   1e-5f

typedef __attribute__((ext_vector_type(16))) __bf16 v16bf;
typedef __attribute__((ext_vector_type(8)))  __bf16 bf16x8;
typedef __attribute__((ext_vector_type(8)))  float  v8f;

union Frag { v16bf v; bf16x8 h[2]; };

// ---------------- wave reduction ----------------
__device__ inline float wave_sum(float v) {
#pragma unroll
    for (int off = 16; off > 0; off >>= 1) v += __shfl_xor(v, off, 32);
    return v;
}

// ---------------- f32 -> bf16 (strided slice) ----------------
__global__ __launch_bounds__(256)
void k_cvt_bf16(const float* __restrict__ src, __bf16* __restrict__ dst,
                long long total, int ncols, int src_ld, int col_off) {
    long long idx = (long long)blockIdx.x * blockDim.x + threadIdx.x;
    if (idx >= total) return;
    long long row = idx / ncols;
    int col = (int)(idx - row * ncols);
    dst[idx] = (__bf16)src[row * (long long)src_ld + col_off + col];
}

// ---------------- embedding gather + LayerNorm ----------------
__global__ __launch_bounds__(256)
void k_embed_ln(const int* __restrict__ src, const float* __restrict__ emb,
                const float* __restrict__ w, const float* __restrict__ b,
                float* __restrict__ out) {
    int row = blockIdx.x * 8 + (threadIdx.x >> 5);
    if (row >= ROWS) return;
    int lane = threadIdx.x & 31;
    const float4* x = (const float4*)(emb + (size_t)src[row] * D_MODEL);
    float4 v[4];
    float s = 0.f;
#pragma unroll
    for (int q = 0; q < 4; ++q) {
        v[q] = x[lane * 4 + q];
        s += v[q].x + v[q].y + v[q].z + v[q].w;
    }
    float mu = wave_sum(s) * (1.f / D_MODEL);
    float ss = 0.f;
#pragma unroll
    for (int q = 0; q < 4; ++q) {
        float dx = v[q].x - mu, dy = v[q].y - mu, dz = v[q].z - mu, dw = v[q].w - mu;
        ss += dx * dx + dy * dy + dz * dz + dw * dw;
    }
    float inv = rsqrtf(wave_sum(ss) * (1.f / D_MODEL) + LN_EPS);
    float* o = out + (size_t)row * D_MODEL;
#pragma unroll
    for (int q = 0; q < 4; ++q) {
        int c = lane * 16 + q * 4;
        o[c + 0] = (v[q].x - mu) * inv * w[c + 0] + b[c + 0];
        o[c + 1] = (v[q].y - mu) * inv * w[c + 1] + b[c + 1];
        o[c + 2] = (v[q].z - mu) * inv * w[c + 2] + b[c + 2];
        o[c + 3] = (v[q].w - mu) * inv * w[c + 3] + b[c + 3];
    }
}

// ------- residual = hidden (+ residual); normed = LN(residual) -> bf16 -------
__global__ __launch_bounds__(256)
void k_add_ln_bf16(const float* __restrict__ hidden, float* __restrict__ residual,
                   const float* __restrict__ w, const float* __restrict__ b,
                   __bf16* __restrict__ normed, int first) {
    int row = blockIdx.x * 8 + (threadIdx.x >> 5);
    if (row >= ROWS) return;
    int lane = threadIdx.x & 31;
    const float4* h = (const float4*)(hidden + (size_t)row * D_MODEL);
    float4* r = (float4*)(residual + (size_t)row * D_MODEL);
    float4 v[4];
    float s = 0.f;
#pragma unroll
    for (int q = 0; q < 4; ++q) {
        float4 hv = h[lane * 4 + q];
        if (!first) {
            float4 rv = r[lane * 4 + q];
            hv.x += rv.x; hv.y += rv.y; hv.z += rv.z; hv.w += rv.w;
        }
        r[lane * 4 + q] = hv;
        v[q] = hv;
        s += hv.x + hv.y + hv.z + hv.w;
    }
    float mu = wave_sum(s) * (1.f / D_MODEL);
    float ss = 0.f;
#pragma unroll
    for (int q = 0; q < 4; ++q) {
        float dx = v[q].x - mu, dy = v[q].y - mu, dz = v[q].z - mu, dw = v[q].w - mu;
        ss += dx * dx + dy * dy + dz * dz + dw * dw;
    }
    float inv = rsqrtf(wave_sum(ss) * (1.f / D_MODEL) + LN_EPS);
    __bf16* o = normed + (size_t)row * D_MODEL;
#pragma unroll
    for (int q = 0; q < 4; ++q) {
        int c = lane * 16 + q * 4;
        o[c + 0] = (__bf16)((v[q].x - mu) * inv * w[c + 0] + b[c + 0]);
        o[c + 1] = (__bf16)((v[q].y - mu) * inv * w[c + 1] + b[c + 1]);
        o[c + 2] = (__bf16)((v[q].z - mu) * inv * w[c + 2] + b[c + 2]);
        o[c + 3] = (__bf16)((v[q].w - mu) * inv * w[c + 3] + b[c + 3]);
    }
}

// ------- final: residual += hidden; out = LN(residual) (f32) -------
__global__ __launch_bounds__(256)
void k_final_ln(const float* __restrict__ hidden, float* __restrict__ residual,
                const float* __restrict__ w, const float* __restrict__ b,
                float* __restrict__ out) {
    int row = blockIdx.x * 8 + (threadIdx.x >> 5);
    if (row >= ROWS) return;
    int lane = threadIdx.x & 31;
    const float4* h = (const float4*)(hidden + (size_t)row * D_MODEL);
    float4* r = (float4*)(residual + (size_t)row * D_MODEL);
    float4 v[4];
    float s = 0.f;
#pragma unroll
    for (int q = 0; q < 4; ++q) {
        float4 hv = h[lane * 4 + q];
        float4 rv = r[lane * 4 + q];
        hv.x += rv.x; hv.y += rv.y; hv.z += rv.z; hv.w += rv.w;
        v[q] = hv;
        s += hv.x + hv.y + hv.z + hv.w;
    }
    float mu = wave_sum(s) * (1.f / D_MODEL);
    float ss = 0.f;
#pragma unroll
    for (int q = 0; q < 4; ++q) {
        float dx = v[q].x - mu, dy = v[q].y - mu, dz = v[q].z - mu, dw = v[q].w - mu;
        ss += dx * dx + dy * dy + dz * dz + dw * dw;
    }
    float inv = rsqrtf(wave_sum(ss) * (1.f / D_MODEL) + LN_EPS);
    float* o = out + (size_t)row * D_MODEL;
#pragma unroll
    for (int q = 0; q < 4; ++q) {
        int c = lane * 16 + q * 4;
        o[c + 0] = (v[q].x - mu) * inv * w[c + 0] + b[c + 0];
        o[c + 1] = (v[q].y - mu) * inv * w[c + 1] + b[c + 1];
        o[c + 2] = (v[q].z - mu) * inv * w[c + 2] + b[c + 2];
        o[c + 3] = (v[q].w - mu) * inv * w[c + 3] + b[c + 3];
    }
}

// ---------------- bf16 WMMA GEMM: C(MxN,f32) = A(MxK,bf16) * W(NxK,bf16)^T ----------------
// block = 128 threads (4 waves); wave tile 64x64; block tile 256(M) x 64(N).
// B tile (64 x 32 bf16, shared by all 4 waves) is staged into LDS with
// GLOBAL_LOAD_ASYNC_TO_LDS_B128 (ASYNCcnt), double-buffered to overlap with WMMA.
// grid = (M/256, N/64); requires K % 32 == 0.
// act: 0 = none, 1 = softplus(acc + bias[n])
#define LDS_ROW 80   // 64B of K data + 16B pad -> no 4-way bank conflict on ds_load_b128

__global__ __launch_bounds__(128)
void k_gemm_bf16(const __bf16* __restrict__ A, int lda,
                 const __bf16* __restrict__ W, int ldw,
                 float* __restrict__ C, int ldc,
                 int K, const float* __restrict__ bias, int act) {
    __shared__ __align__(16) char shB[2][64 * LDS_ROW];   // 10 KB

    const int lane   = threadIdx.x & 31;
    const int wave   = threadIdx.x >> 5;
    const int lane15 = lane & 15;
    const int half   = lane >> 4;                 // 0 or 1
    const int tileM  = blockIdx.x * 256 + wave * 64;
    const int tileN  = blockIdx.y * 64;

    // async-copy assignment: thread t moves 32 bytes of row (t>>1), byte offset (t&1)*32
    const int crow = threadIdx.x >> 1;
    const int cpos = (threadIdx.x & 1) * 32;
    const __bf16* wrow = W + (size_t)(tileN + crow) * ldw;   // + kk elements later

    auto async_copy = [&](int kk, int buf) {
        unsigned      l0 = (unsigned)(size_t)&shB[buf][crow * LDS_ROW + cpos];
        unsigned long long g0 = (unsigned long long)(size_t)(wrow + kk) + (unsigned)cpos;
        asm volatile("global_load_async_to_lds_b128 %0, %1, off\n\t"
                     "global_load_async_to_lds_b128 %2, %3, off"
                     :: "v"(l0), "v"(g0), "v"(l0 + 16u), "v"(g0 + 16ull)
                     : "memory");
    };

    v8f acc[4][4] = {};

    async_copy(0, 0);
    asm volatile("s_wait_asynccnt 0" ::: "memory");
    __syncthreads();

    int it = 0;
    for (int kk = 0; kk < K; kk += 32, ++it) {
        const int cur = it & 1;
        if (kk + 32 < K) {
            async_copy(kk + 32, cur ^ 1);
            __builtin_prefetch(A + (size_t)(tileM + lane15) * lda + kk + 32, 0, 3);
        }

        Frag a[4], w[4];
#pragma unroll
        for (int i = 0; i < 4; ++i) {
            // A 16x32 bf16 layout: lanes 0-15 hold K{kk..kk+7, kk+16..kk+23},
            // lanes 16-31 hold K{kk+8..kk+15, kk+24..kk+31}
            const __bf16* ap = A + (size_t)(tileM + i * 16 + lane15) * lda + kk + half * 8;
            a[i].h[0] = *(const bf16x8*)ap;
            a[i].h[1] = *(const bf16x8*)(ap + 16);
        }
#pragma unroll
        for (int j = 0; j < 4; ++j) {
            // B 32x16 bf16 layout from LDS: lane holds col n=lane15, K = kk + half*8 + {0..7, 16..23}
            const char* bp = &shB[cur][(j * 16 + lane15) * LDS_ROW + half * 16];
            w[j].h[0] = *(const bf16x8*)bp;
            w[j].h[1] = *(const bf16x8*)(bp + 32);
        }
#pragma unroll
        for (int i = 0; i < 4; ++i)
#pragma unroll
            for (int j = 0; j < 4; ++j)
                acc[i][j] = __builtin_amdgcn_wmma_f32_16x16x32_bf16(
                    false, a[i].v, false, w[j].v, (short)0, acc[i][j], false, false);

        asm volatile("s_wait_asynccnt 0" ::: "memory");
        __syncthreads();
    }

#pragma unroll
    for (int i = 0; i < 4; ++i) {
#pragma unroll
        for (int j = 0; j < 4; ++j) {
            int n = tileN + j * 16 + lane15;
            float bv = bias ? bias[n] : 0.f;
#pragma unroll
            for (int r = 0; r < 8; ++r) {
                int m = tileM + i * 16 + half * 8 + r;
                float v = acc[i][j][r] + bv;
                if (act == 1) v = (v > 20.f) ? v : __logf(1.f + __expf(v));
                C[(size_t)m * ldc + n] = v;
            }
        }
    }
}

// ---------------- depthwise causal conv (4 taps) + SiLU -> bf16 ----------------
__global__ __launch_bounds__(256)
void k_conv_silu(const float* __restrict__ xz, const float* __restrict__ cw,
                 const float* __restrict__ cb, __bf16* __restrict__ xout) {
    long long idx = (long long)blockIdx.x * blockDim.x + threadIdx.x;
    if (idx >= (long long)ROWS * D_INNER) return;
    int d = (int)(idx & (D_INNER - 1));
    long long t = idx >> 10;           // token index
    int l = (int)(t & (SEQ - 1));
    long long b = t >> 11;
    float acc = cb[d];
#pragma unroll
    for (int k = 0; k < D_CONV; ++k) {
        int ls = l - (D_CONV - 1) + k;
        if (ls >= 0)
            acc += cw[d * D_CONV + k] * xz[((b * SEQ + ls) * (long long)XZ_N) + d];
    }
    float s = acc / (1.f + __expf(-acc));   // SiLU
    xout[idx] = (__bf16)s;
}

// ---------------- fused selective scan + D-skip + SiLU(z) gate -> bf16 ----------------
__global__ __launch_bounds__(256)
void k_scan(const float* __restrict__ dt, const float* __restrict__ xdbl,
            const __bf16* __restrict__ xconv, const float* __restrict__ xz,
            const float* __restrict__ A_log, const float* __restrict__ Dskip,
            __bf16* __restrict__ yout) {
    int gid = blockIdx.x * blockDim.x + threadIdx.x;   // 0 .. B*D_INNER-1
    if (gid >= B_SZ * D_INNER) return;
    int d = gid & (D_INNER - 1);
    int b = gid >> 10;

    float A[D_STATE], st[D_STATE];
#pragma unroll
    for (int s = 0; s < D_STATE; ++s) {
        A[s] = -__expf(A_log[d * D_STATE + s]);
        st[s] = 0.f;
    }
    float Dd = Dskip[d];

    for (int l = 0; l < SEQ; ++l) {
        size_t row = (size_t)b * SEQ + l;
        float dtv = dt[row * D_INNER + d];
        float xv  = (float)xconv[row * D_INNER + d];
        float zv  = xz[row * XZ_N + D_INNER + d];
        const float4* B4 = (const float4*)(xdbl + row * XDBL_N + DT_RANK);
        const float4* C4 = B4 + 4;
        float Bv[D_STATE], Cv[D_STATE];
#pragma unroll
        for (int q = 0; q < 4; ++q) {
            *(float4*)&Bv[q * 4] = B4[q];
            *(float4*)&Cv[q * 4] = C4[q];
        }
        float dx = dtv * xv;
        float y = 0.f;
#pragma unroll
        for (int s = 0; s < D_STATE; ++s) {
            float dA = __expf(dtv * A[s]);
            st[s] = dA * st[s] + dx * Bv[s];
            y += st[s] * Cv[s];
        }
        y += xv * Dd;
        float g = zv / (1.f + __expf(-zv));
        yout[row * D_INNER + d] = (__bf16)(y * g);
    }
}

// ---------------- host orchestration ----------------
static inline void* wsTake(char*& p, size_t bytes) {
    void* r = (void*)p;
    p += (bytes + 255) & ~(size_t)255;
    return r;
}

extern "C" void kernel_launch(void* const* d_in, const int* in_sizes, int n_in,
                              void* d_out, int out_size, void* d_ws, size_t ws_size,
                              hipStream_t stream) {
    const int*   src        = (const int*)  d_in[0];
    // d_in[1] = src_key_padding_mask (unused by reference)
    const float* emb_w      = (const float*)d_in[2];
    const float* enc_ln_w   = (const float*)d_in[3];
    const float* enc_ln_b   = (const float*)d_in[4];
    const float* ln_w       = (const float*)d_in[5];
    const float* ln_b       = (const float*)d_in[6];
    const float* in_proj_w  = (const float*)d_in[7];
    const float* conv_w     = (const float*)d_in[8];
    const float* conv_b     = (const float*)d_in[9];
    const float* x_proj_w   = (const float*)d_in[10];
    const float* dt_proj_w  = (const float*)d_in[11];
    const float* dt_proj_b  = (const float*)d_in[12];
    const float* A_log      = (const float*)d_in[13];
    const float* D_skip     = (const float*)d_in[14];
    const float* out_proj_w = (const float*)d_in[15];
    const float* normf_w    = (const float*)d_in[16];
    const float* normf_b    = (const float*)d_in[17];
    float* out = (float*)d_out;

    char* p = (char*)d_ws;
    __bf16* wt_in  = (__bf16*)wsTake(p, (size_t)NLAYERS * XZ_N * D_MODEL * 2);
    __bf16* wt_x   = (__bf16*)wsTake(p, (size_t)NLAYERS * XDBL_N * D_INNER * 2);
    __bf16* wt_dt  = (__bf16*)wsTake(p, (size_t)NLAYERS * D_INNER * DT_RANK * 2);
    __bf16* wt_out = (__bf16*)wsTake(p, (size_t)NLAYERS * D_MODEL * D_INNER * 2);
    float*  hidden   = (float*) wsTake(p, (size_t)ROWS * D_MODEL * 4);
    float*  residual = (float*) wsTake(p, (size_t)ROWS * D_MODEL * 4);
    __bf16* normed   = (__bf16*)wsTake(p, (size_t)ROWS * D_MODEL * 2);
    float*  xz       = (float*) wsTake(p, (size_t)ROWS * XZ_N * 4);
    __bf16* xconv    = (__bf16*)wsTake(p, (size_t)ROWS * D_INNER * 2);
    float*  xdbl     = (float*) wsTake(p, (size_t)ROWS * XDBL_N * 4);
    __bf16* dtA      = (__bf16*)wsTake(p, (size_t)ROWS * DT_RANK * 2);
    float*  dtbuf    = (float*) wsTake(p, (size_t)ROWS * D_INNER * 4);
    __bf16* ybf      = (__bf16*)wsTake(p, (size_t)ROWS * D_INNER * 2);
    (void)ws_size; (void)in_sizes; (void)n_in; (void)out_size;

    auto cvt = [&](const float* s, __bf16* dst, long long total, int ncols, int ld, int off) {
        int blocks = (int)((total + 255) / 256);
        k_cvt_bf16<<<blocks, 256, 0, stream>>>(s, dst, total, ncols, ld, off);
    };

    // one-shot weight converts (whole (NLAYERS, ...) tensors, contiguous)
    cvt(in_proj_w,  wt_in,  (long long)NLAYERS * XZ_N * D_MODEL, D_MODEL, D_MODEL, 0);
    cvt(x_proj_w,   wt_x,   (long long)NLAYERS * XDBL_N * D_INNER, D_INNER, D_INNER, 0);
    cvt(dt_proj_w,  wt_dt,  (long long)NLAYERS * D_INNER * DT_RANK, DT_RANK, DT_RANK, 0);
    cvt(out_proj_w, wt_out, (long long)NLAYERS * D_MODEL * D_INNER, D_INNER, D_INNER, 0);

    // embedding + encoder LayerNorm -> hidden
    k_embed_ln<<<ROWS / 8, 256, 0, stream>>>(src, emb_w, enc_ln_w, enc_ln_b, hidden);

    auto gemm = [&](const __bf16* A, int lda, const __bf16* W, int ldw,
                    float* C, int ldc, int M, int N, int K,
                    const float* bias, int act) {
        dim3 g(M / 256, N / 64);
        k_gemm_bf16<<<g, 128, 0, stream>>>(A, lda, W, ldw, C, ldc, K, bias, act);
    };

    for (int i = 0; i < NLAYERS; ++i) {
        // residual accumulate + pre-LN (bf16 out)
        k_add_ln_bf16<<<ROWS / 8, 256, 0, stream>>>(hidden, residual,
                                                    ln_w + i * D_MODEL, ln_b + i * D_MODEL,
                                                    normed, i == 0 ? 1 : 0);
        // xz = normed @ in_proj^T   (8192 x 2048 x 512)
        gemm(normed, D_MODEL, wt_in + (size_t)i * XZ_N * D_MODEL, D_MODEL,
             xz, XZ_N, ROWS, XZ_N, D_MODEL, nullptr, 0);
        // depthwise conv + SiLU -> bf16
        {
            long long total = (long long)ROWS * D_INNER;
            k_conv_silu<<<(int)((total + 255) / 256), 256, 0, stream>>>(
                xz, conv_w + (size_t)i * D_INNER * D_CONV, conv_b + (size_t)i * D_INNER, xconv);
        }
        // xdbl = x @ x_proj^T   (8192 x 64 x 1024)
        gemm(xconv, D_INNER, wt_x + (size_t)i * XDBL_N * D_INNER, D_INNER,
             xdbl, XDBL_N, ROWS, XDBL_N, D_INNER, nullptr, 0);
        // bf16 slice of dt-rank columns
        cvt(xdbl, dtA, (long long)ROWS * DT_RANK, DT_RANK, XDBL_N, 0);
        // dt = softplus(dtA @ dt_proj^T + dt_b)   (8192 x 1024 x 32)
        gemm(dtA, DT_RANK, wt_dt + (size_t)i * D_INNER * DT_RANK, DT_RANK,
             dtbuf, D_INNER, ROWS, D_INNER, DT_RANK, dt_proj_b + (size_t)i * D_INNER, 1);
        // selective scan + gate -> ybf
        k_scan<<<(B_SZ * D_INNER) / 256, 256, 0, stream>>>(
            dtbuf, xdbl, xconv, xz,
            A_log + (size_t)i * D_INNER * D_STATE, D_skip + (size_t)i * D_INNER, ybf);
        // hidden = y @ out_proj^T   (8192 x 512 x 1024)
        gemm(ybf, D_INNER, wt_out + (size_t)i * D_MODEL * D_INNER, D_INNER,
             hidden, D_MODEL, ROWS, D_MODEL, D_INNER, nullptr, 0);
    }

    // final residual + LayerNorm -> d_out (f32)
    k_final_ln<<<ROWS / 8, 256, 0, stream>>>(hidden, residual, normf_w, normf_b, out);
}
